// PatchedLlamaAttention_63359357551225
// MI455X (gfx1250) — matrix operand order repaired and checked by reading, not compiled, not run on hardware
//
#include <hip/hip_runtime.h>
#include <hip/hip_bf16.h>
#include <cstdint>
#include <cstddef>

typedef _Float16 v16h __attribute__((ext_vector_type(16)));
typedef _Float16 v8h  __attribute__((ext_vector_type(8)));
typedef float    v8f  __attribute__((ext_vector_type(8)));

#define S_LEN   4096
#define D_MODEL 2048
#define N_H     16
#define N_KV    4
#define HD      128
#define KVD     (N_KV * HD)      /* 512  */
#define GROUPS  (N_H / N_KV)     /* 4    */

union F16x16 { v16h v; v8h h[2]; };

#if __has_builtin(__builtin_amdgcn_global_load_async_to_lds_b128)
#define ASYNC_LDS 1
#else
#define ASYNC_LDS 0
#endif

static __device__ __forceinline__ v8f wmma_f32_16x16x32_f16(v16h a, v16h b, v8f c) {
  return __builtin_amdgcn_wmma_f32_16x16x32_f16(false, a, false, b, (short)0, c, false, false);
}

// 16-byte global -> LDS copy: async DMA path on gfx1250 if the builtin exists,
// otherwise a plain vector copy through VGPRs.
#if ASYNC_LDS
typedef int ai4 __attribute__((vector_size(16)));
static __device__ __forceinline__ void async_copy16(const _Float16* g, _Float16* l) {
  __builtin_amdgcn_global_load_async_to_lds_b128(
      (__attribute__((address_space(1))) ai4*)const_cast<_Float16*>(g),
      (__attribute__((address_space(3))) ai4*)l, 0, 0);
}
#else
static __device__ __forceinline__ void async_copy16(const _Float16* g, _Float16* l) {
  *(v8h*)l = *(const v8h*)g;
}
#endif

static __device__ __forceinline__ void wait_async_all() {
#if ASYNC_LDS
#if __has_builtin(__builtin_amdgcn_s_wait_asynccnt)
  __builtin_amdgcn_s_wait_asynccnt(0);
#else
  asm volatile("s_wait_asynccnt 0x0" ::: "memory");
#endif
#endif
}

// ---------------------------------------------------------------------------
// Elementwise f32 -> f16 cast
// ---------------------------------------------------------------------------
__global__ void cast_f32_f16_kernel(const float* __restrict__ src,
                                    _Float16* __restrict__ dst, int n) {
  int i = blockIdx.x * blockDim.x + threadIdx.x;
  if (i < n) dst[i] = (_Float16)src[i];
}

// ---------------------------------------------------------------------------
// RoPE (interleaved-pair convention) fused with f32 -> f16.
// X/out layout: [S][NH*HD]; one thread per (s, h, d<64) pair.
// ---------------------------------------------------------------------------
__global__ void rope_f32_f16_kernel(const float* __restrict__ X,
                                    const float* __restrict__ cs,
                                    const float* __restrict__ sn,
                                    _Float16* __restrict__ out, int NH) {
  int i = blockIdx.x * blockDim.x + threadIdx.x;
  int d = i & 63;
  int h = (i >> 6) % NH;
  int s = i / (NH * 64);
  const float* xp = X + ((size_t)s * NH + h) * HD;
  float x1 = xp[2 * d];
  float x2 = xp[2 * d + 1];
  float c  = cs[s * 64 + d];
  float si = sn[s * 64 + d];
  _Float16* op = out + ((size_t)s * NH + h) * HD;
  op[d]      = (_Float16)(x1 * c - x2 * si);
  op[d + 64] = (_Float16)(x2 * c + x1 * si);
}

// ---------------------------------------------------------------------------
// Tiled GEMM: C(f32, MxN) = A(f16, MxK) * B(f16, KxN)
// Block: 256 threads (8 waves), 128x128 tile, K-step 32, double-buffered LDS.
// Wave (wm,wn) grid 4x2: each wave computes 32 rows x 64 cols =
// 2x4 WMMA accumulators -> 8 v_wmma per wave per K-step from 6 fragment loads.
// A tile staged via async-to-LDS; B tile staged transposed ([n][k]) so
// B fragments are two 16B ds loads per lane (k = 8h.., 16+8h..).
// Requires M%128==0, N%128==0, K%32==0 (true for all uses here).
// ---------------------------------------------------------------------------
#define SAST 48
#define SBST 48

__global__ __launch_bounds__(256) void gemm_f16_kernel(
    const _Float16* __restrict__ A, const _Float16* __restrict__ B,
    float* __restrict__ C, int M, int N, int K) {
  __shared__ __align__(32) _Float16 sA[2][128][SAST];    // [buf][m][k]
  __shared__ __align__(32) _Float16 sBT[2][128][SBST];   // [buf][n][k]
  const int tid    = threadIdx.x;
  const int wave   = tid >> 5;
  const int lane   = tid & 31;
  const int laneN  = lane & 15;
  const int laneHi = lane >> 4;
  const int wm = (wave & 3) * 32;
  const int wn = (wave >> 2) * 64;
  const int m0 = blockIdx.y * 128;
  const int n0 = blockIdx.x * 128;

  // staging roles
  const int arow = tid >> 1;            // 0..127
  const int aseg = (tid & 1) * 16;      // 0 or 16 halves
  const int bkk  = tid >> 3;            // 0..31  (k row)
  const int bnn  = (tid & 7) * 16;      // 0..112 (n segment)

  v8f acc[2][4] = {};

  // ---- prologue: stage buffer 0 ----
  {
    const _Float16* ga = A + (size_t)(m0 + arow) * K + aseg;
    async_copy16(ga, &sA[0][arow][aseg]);
    async_copy16(ga + 8, &sA[0][arow][aseg + 8]);
    const v8h* src = (const v8h*)(B + (size_t)bkk * N + n0 + bnn);
    v8h b0 = src[0], b1 = src[1];
#pragma unroll
    for (int i = 0; i < 8; ++i) {
      sBT[0][bnn + i][bkk]     = b0[i];
      sBT[0][bnn + 8 + i][bkk] = b1[i];
    }
  }
  wait_async_all();
  __syncthreads();

  const int nk = K / 32;
  for (int ki = 0; ki < nk; ++ki) {
    const int cur = ki & 1;
    const int nxt = cur ^ 1;
    const bool has_next = (ki + 1) < nk;
    v8h b0 = {}, b1 = {};
    if (has_next) { // kick off staging of the next K-tile into the other buffer
      const int k0n = (ki + 1) * 32;
      const _Float16* ga = A + (size_t)(m0 + arow) * K + k0n + aseg;
      async_copy16(ga, &sA[nxt][arow][aseg]);
      async_copy16(ga + 8, &sA[nxt][arow][aseg + 8]);
      const v8h* src = (const v8h*)(B + (size_t)(k0n + bkk) * N + n0 + bnn);
      b0 = src[0];
      b1 = src[1];
    }

    // ---- compute on current buffer ----
    F16x16 af[2];
#pragma unroll
    for (int ms = 0; ms < 2; ++ms) {
      af[ms].h[0] = *(const v8h*)&sA[cur][wm + ms * 16 + laneN][laneHi * 8];
      af[ms].h[1] = *(const v8h*)&sA[cur][wm + ms * 16 + laneN][16 + laneHi * 8];
    }
#pragma unroll
    for (int ns = 0; ns < 4; ++ns) {
      F16x16 bfu;
      bfu.h[0] = *(const v8h*)&sBT[cur][wn + ns * 16 + laneN][laneHi * 8];
      bfu.h[1] = *(const v8h*)&sBT[cur][wn + ns * 16 + laneN][16 + laneHi * 8];
      v16h bf = bfu.v;
#pragma unroll
      for (int ms = 0; ms < 2; ++ms)
        acc[ms][ns] = wmma_f32_16x16x32_f16(af[ms].v, bf, acc[ms][ns]);
    }

    if (has_next) { // transpose-store B for the next tile
#pragma unroll
      for (int i = 0; i < 8; ++i) {
        sBT[nxt][bnn + i][bkk]     = b0[i];
        sBT[nxt][bnn + 8 + i][bkk] = b1[i];
      }
    }
    wait_async_all();
    __syncthreads();
  }

#pragma unroll
  for (int ms = 0; ms < 2; ++ms)
#pragma unroll
    for (int ns = 0; ns < 4; ++ns)
#pragma unroll
      for (int vg = 0; vg < 8; ++vg) {
        int m = m0 + wm + ms * 16 + vg + 8 * laneHi;
        int n = n0 + wn + ns * 16 + laneN;
        C[(size_t)m * N + n] = acc[ms][ns][vg];
      }
}

// ---------------------------------------------------------------------------
// Flash attention (causal, GQA). Block = 64 query rows x 1 head, 4 waves.
// Q/K/V f16: Q [S][H*HD], K/V [S][HKV*HD]. Output AO f16 [S][H*HD].
// ---------------------------------------------------------------------------
#define SKST 136
#define SVST 48
#define SPST 48

__global__ __launch_bounds__(128) void attn_kernel(
    const _Float16* __restrict__ Q, const _Float16* __restrict__ Kf,
    const _Float16* __restrict__ Vf, _Float16* __restrict__ O) {
  __shared__ __align__(32) _Float16 sK[32][SKST];      // [key][hd]
  __shared__ __align__(32) _Float16 sVt[HD][SVST];     // [hd][key]
  __shared__ __align__(32) _Float16 sP[4][16][SPST];   // per-wave P tile

  const int tid    = threadIdx.x;
  const int wave   = tid >> 5;
  const int lane   = tid & 31;
  const int laneN  = lane & 15;
  const int laneHi = lane >> 4;
  const int head   = blockIdx.y;
  const int kvh    = head / GROUPS;
  const int qblk   = blockIdx.x * 64;
  const int qb     = qblk + wave * 16;

  // Q fragments for the 4 hd K-steps, kept in registers for the whole pass
  F16x16 qf[4];
  {
    const _Float16* qp = Q + (size_t)(qb + laneN) * D_MODEL + head * HD;
#pragma unroll
    for (int kk = 0; kk < 4; ++kk) {
      qf[kk].h[0] = *(const v8h*)(qp + kk * 32 + laneHi * 8);
      qf[kk].h[1] = *(const v8h*)(qp + kk * 32 + 16 + laneHi * 8);
    }
  }

  v8f o[8] = {};
  float mrow[8], lrow[8];
#pragma unroll
  for (int vg = 0; vg < 8; ++vg) { mrow[vg] = -1e30f; lrow[vg] = 0.0f; }

  const int skey = tid >> 2;             // staging: key row 0..31
  const int sseg = (tid & 3) * 32;       // staging: 32-half segment

  const int nkb = qblk / 32 + 2;
  for (int kb = 0; kb < nkb; ++kb) {
    const int kbs = kb * 32;
    { // stage K tile (async path) and transposed V tile
      const _Float16* kg = Kf + (size_t)(kbs + skey) * KVD + kvh * HD + sseg;
      _Float16* kl = &sK[skey][sseg];
      async_copy16(kg, kl);
      async_copy16(kg + 8, kl + 8);
      async_copy16(kg + 16, kl + 16);
      async_copy16(kg + 24, kl + 24);

      const v8h* vs = (const v8h*)(Vf + (size_t)(kbs + skey) * KVD + kvh * HD + sseg);
      v8h v0 = vs[0], v1 = vs[1], v2 = vs[2], v3 = vs[3];
#pragma unroll
      for (int i = 0; i < 8; ++i) {
        sVt[sseg + i][skey]      = v0[i];
        sVt[sseg + 8 + i][skey]  = v1[i];
        sVt[sseg + 16 + i][skey] = v2[i];
        sVt[sseg + 24 + i][skey] = v3[i];
      }
      if (kb + 1 < nkb) { // prefetch next key block (global_prefetch_b8)
        __builtin_prefetch(Kf + (size_t)(kbs + 32 + skey) * KVD + kvh * HD + sseg, 0, 0);
        __builtin_prefetch(Vf + (size_t)(kbs + 32 + skey) * KVD + kvh * HD + sseg, 0, 0);
      }
    }
    wait_async_all();
    __syncthreads();

    // S = Q * K^T (16 q-rows x 32 keys)
    v8f s[2] = {};
#pragma unroll
    for (int ksub = 0; ksub < 2; ++ksub)
#pragma unroll
      for (int kk = 0; kk < 4; ++kk) {
        F16x16 bf;
        bf.h[0] = *(const v8h*)&sK[ksub * 16 + laneN][kk * 32 + laneHi * 8];
        bf.h[1] = *(const v8h*)&sK[ksub * 16 + laneN][kk * 32 + 16 + laneHi * 8];
        s[ksub] = wmma_f32_16x16x32_f16(qf[kk].v, bf.v, s[ksub]);
      }

    // online softmax (rows split across lane halves; xor masks 1/2/4/8 stay
    // inside each 16-lane group)
    const float scale = 0.08838834764831845f;  // 1/sqrt(128)
#pragma unroll
    for (int vg = 0; vg < 8; ++vg) {
      int qrow = qb + vg + 8 * laneHi;
      float sv0 = s[0][vg] * scale + ((kbs + laneN)      > qrow ? -1e9f : 0.0f);
      float sv1 = s[1][vg] * scale + ((kbs + 16 + laneN) > qrow ? -1e9f : 0.0f);
      float sm = fmaxf(sv0, sv1);
      sm = fmaxf(sm, __shfl_xor(sm, 8));
      sm = fmaxf(sm, __shfl_xor(sm, 4));
      sm = fmaxf(sm, __shfl_xor(sm, 2));
      sm = fmaxf(sm, __shfl_xor(sm, 1));
      float mnew  = fmaxf(mrow[vg], sm);
      float alpha = __expf(mrow[vg] - mnew);
      float p0 = __expf(sv0 - mnew);
      float p1 = __expf(sv1 - mnew);
      float ps = p0 + p1;
      ps += __shfl_xor(ps, 8);
      ps += __shfl_xor(ps, 4);
      ps += __shfl_xor(ps, 2);
      ps += __shfl_xor(ps, 1);
      lrow[vg] = lrow[vg] * alpha + ps;
      mrow[vg] = mnew;
#pragma unroll
      for (int nsub = 0; nsub < 8; ++nsub) o[nsub][vg] *= alpha;
      sP[wave][vg + 8 * laneHi][laneN]      = (_Float16)p0;
      sP[wave][vg + 8 * laneHi][16 + laneN] = (_Float16)p1;
    }
    __syncthreads();

    // O += P(16x32) * V(32x128)
    F16x16 pf;
    pf.h[0] = *(const v8h*)&sP[wave][laneN][laneHi * 8];
    pf.h[1] = *(const v8h*)&sP[wave][laneN][16 + laneHi * 8];
#pragma unroll
    for (int nsub = 0; nsub < 8; ++nsub) {
      F16x16 bfu;
      bfu.h[0] = *(const v8h*)&sVt[nsub * 16 + laneN][laneHi * 8];
      bfu.h[1] = *(const v8h*)&sVt[nsub * 16 + laneN][16 + laneHi * 8];
      v16h bf = bfu.v;
      o[nsub] = wmma_f32_16x16x32_f16(pf.v, bf, o[nsub]);
    }
    __syncthreads();
  }

#pragma unroll
  for (int nsub = 0; nsub < 8; ++nsub)
#pragma unroll
    for (int vg = 0; vg < 8; ++vg) {
      int qrow = qb + vg + 8 * laneHi;
      O[(size_t)qrow * D_MODEL + head * HD + nsub * 16 + laneN] =
          (_Float16)(o[nsub][vg] / lrow[vg]);
    }
}

// ---------------------------------------------------------------------------
// Launch
// ---------------------------------------------------------------------------
extern "C" void kernel_launch(void* const* d_in, const int* in_sizes, int n_in,
                              void* d_out, int out_size, void* d_ws, size_t ws_size,
                              hipStream_t stream) {
  (void)in_sizes; (void)n_in; (void)out_size; (void)ws_size;
  const float* hs = (const float*)d_in[0];
  // d_in[1] = attention_mask (causal -1e9): implemented analytically
  const float* cs = (const float*)d_in[2];
  const float* sn = (const float*)d_in[3];
  const float* Wq = (const float*)d_in[4];
  const float* Wk = (const float*)d_in[5];
  const float* Wv = (const float*)d_in[6];
  const float* Wo = (const float*)d_in[7];
  float* out = (float*)d_out;

  char* w = (char*)d_ws;
  size_t off = 0;
  auto alloc = [&](size_t bytes) -> char* {
    char* p = w + off;
    off += (bytes + 255) & ~(size_t)255;
    return p;
  };
  _Float16* hsb  = (_Float16*)alloc((size_t)S_LEN * D_MODEL * 2);
  _Float16* Wq16 = (_Float16*)alloc((size_t)D_MODEL * D_MODEL * 2);
  _Float16* Wk16 = (_Float16*)alloc((size_t)D_MODEL * KVD * 2);
  _Float16* Wv16 = (_Float16*)alloc((size_t)D_MODEL * KVD * 2);
  _Float16* Wo16 = (_Float16*)alloc((size_t)D_MODEL * D_MODEL * 2);
  float*    Qf   = (float*)   alloc((size_t)S_LEN * D_MODEL * 4);
  float*    Kf   = (float*)   alloc((size_t)S_LEN * KVD * 4);
  float*    Vf   = (float*)   alloc((size_t)S_LEN * KVD * 4);
  _Float16* Q16  = (_Float16*)alloc((size_t)S_LEN * D_MODEL * 2);
  _Float16* K16  = (_Float16*)alloc((size_t)S_LEN * KVD * 2);
  _Float16* V16  = (_Float16*)alloc((size_t)S_LEN * KVD * 2);
  _Float16* AO16 = (_Float16*)alloc((size_t)S_LEN * D_MODEL * 2);

  auto cast = [&](const float* s, _Float16* d, size_t n) {
    cast_f32_f16_kernel<<<(unsigned)((n + 255) / 256), 256, 0, stream>>>(s, d, (int)n);
  };
  cast(hs, hsb, (size_t)S_LEN * D_MODEL);
  cast(Wq, Wq16, (size_t)D_MODEL * D_MODEL);
  cast(Wk, Wk16, (size_t)D_MODEL * KVD);
  cast(Wv, Wv16, (size_t)D_MODEL * KVD);
  cast(Wo, Wo16, (size_t)D_MODEL * D_MODEL);

  dim3 blk(256);
  gemm_f16_kernel<<<dim3(D_MODEL / 128, S_LEN / 128), blk, 0, stream>>>(
      hsb, Wq16, Qf, S_LEN, D_MODEL, D_MODEL);
  gemm_f16_kernel<<<dim3(KVD / 128, S_LEN / 128), blk, 0, stream>>>(
      hsb, Wk16, Kf, S_LEN, KVD, D_MODEL);
  gemm_f16_kernel<<<dim3(KVD / 128, S_LEN / 128), blk, 0, stream>>>(
      hsb, Wv16, Vf, S_LEN, KVD, D_MODEL);

  rope_f32_f16_kernel<<<(S_LEN * N_H * 64) / 256, 256, 0, stream>>>(Qf, cs, sn, Q16, N_H);
  rope_f32_f16_kernel<<<(S_LEN * N_KV * 64) / 256, 256, 0, stream>>>(Kf, cs, sn, K16, N_KV);
  cast(Vf, V16, (size_t)S_LEN * KVD);

  attn_kernel<<<dim3(S_LEN / 64, N_H), dim3(128), 0, stream>>>(Q16, K16, V16, AO16);

  gemm_f16_kernel<<<dim3(D_MODEL / 128, S_LEN / 128), blk, 0, stream>>>(
      AO16, Wo16, out, S_LEN, D_MODEL, D_MODEL);
}
